// tensor_field_net_4174708212032
// MI455X (gfx1250) — matrix operand order repaired and checked by reading, not compiled, not run on hardware
//
#include <hip/hip_runtime.h>
#include <hip/hip_bf16.h>

typedef __attribute__((ext_vector_type(16))) _Float16 v16h;
typedef __attribute__((ext_vector_type(8)))  float    v8f;

#define NPATH 11
#define FEAT 144
#define TILE 16

#if defined(__has_builtin)
#if __has_builtin(__builtin_amdgcn_global_load_async_to_lds_b128) && \
    __has_builtin(__builtin_amdgcn_s_wait_asynccnt)
#define USE_ASYNC_LDS 1
#endif
#endif

#ifdef USE_ASYNC_LDS
typedef int vi4 __attribute__((__vector_size__(4 * sizeof(int))));
typedef __attribute__((address_space(1))) vi4* gptr4;
typedef __attribute__((address_space(3))) vi4* lptr4;
#endif

__constant__ int cL1[NPATH] = {0,1,2,0,1,1,2,0,1,2,2};
__constant__ int cL2[NPATH] = {0,1,2,1,0,2,1,2,1,0,2};
__constant__ int cL3[NPATH] = {0,0,0,1,1,1,1,2,2,2,2};
__constant__ int cOFF[3]    = {0,16,64};

// ---------------- Wigner 3j precompute (device, double precision) ----------------

struct cplx { double re, im; };
__device__ inline cplx cmul(cplx a, cplx b){ return {a.re*b.re - a.im*b.im, a.re*b.im + a.im*b.re}; }

__device__ inline double dfact(int n){ double r = 1.0; for (int i = 2; i <= n; ++i) r *= (double)i; return r; }
__device__ inline int imax2(int a,int b){ return a>b?a:b; }
__device__ inline int imin2(int a,int b){ return a<b?a:b; }

__device__ double su2_cg(int j1,int m1,int j2,int m2,int j3,int m3){
    if (m3 != m1 + m2) return 0.0;
    int vmin = imax2(imax2(-j1+j2+m3, -j1+m1), 0);
    int vmax = imin2(imin2(j2+j3+m1, j3-j1+j2), j3+m3);
    double C = sqrt((2.0*j3+1.0)*dfact(j3+j1-j2)*dfact(j3-j1+j2)*dfact(j1+j2-j3)
                    *dfact(j3+m3)*dfact(j3-m3)
                    /(dfact(j1+j2+j3+1)*dfact(j1-m1)*dfact(j1+m1)*dfact(j2-m2)*dfact(j2+m2)));
    double S = 0.0;
    for (int v = vmin; v <= vmax; ++v) {
        double sgn = ((v + j2 + m2) & 1) ? -1.0 : 1.0;
        S += sgn * dfact(j2+j3+m1-v) * dfact(j1-m1+v)
             / (dfact(v) * dfact(j3-j1+j2-v) * dfact(j3+m3-v) * dfact(v+j1-j2-m3));
    }
    return C * S;
}

// complex->real SH change of basis (e3nn convention), element [r][c], times (-i)^l
__device__ cplx qentry(int l, int r, int c){
    int m = r - l;
    const double s2 = 0.7071067811865475244;
    cplx v = {0.0, 0.0};
    if (m < 0) {
        int am = -m;
        if (c == l + am) v = {s2, 0.0};
        if (c == l - am) v = {0.0, -s2};
    } else if (m == 0) {
        if (c == l) v = {1.0, 0.0};
    } else {
        double sgn = (m & 1) ? -1.0 : 1.0;
        if (c == l + m) v = {sgn * s2, 0.0};
        if (c == l - m) v = {0.0, sgn * s2};
    }
    if (l == 1)      v = { v.im, -v.re};   // * (-i)
    else if (l == 2) v = {-v.re, -v.im};   // * (-1)
    return v;
}

__global__ void w3j_kernel(float* __restrict__ w3j){
    int p = threadIdx.x;
    if (p >= NPATH) return;
    int l1 = cL1[p], l2 = cL2[p], l3 = cL3[p];
    int d1 = 2*l1+1, d2 = 2*l2+1, d3 = 2*l3+1;
    double Ct[5][5][5];
    for (int i = 0; i < d1; ++i)
        for (int k = 0; k < d2; ++k)
            for (int n = 0; n < d3; ++n)
                Ct[i][k][n] = su2_cg(l1, i-l1, l2, k-l2, l3, n-l3);
    float* T = w3j + p * 128;
    double nrm = 0.0;
    for (int j = 0; j < d1; ++j)
      for (int l = 0; l < d2; ++l)
        for (int m = 0; m < d3; ++m) {
            cplx s = {0.0, 0.0};
            for (int i = 0; i < d1; ++i) {
                cplx q1 = qentry(l1, i, j);
                if (q1.re == 0.0 && q1.im == 0.0) continue;
                for (int k = 0; k < d2; ++k) {
                    cplx q2 = qentry(l2, k, l);
                    if (q2.re == 0.0 && q2.im == 0.0) continue;
                    cplx q12 = cmul(q1, q2);
                    for (int n = 0; n < d3; ++n) {
                        double c = Ct[i][k][n];
                        if (c == 0.0) continue;
                        cplx q3 = qentry(l3, n, m);
                        q3.im = -q3.im;                 // conj
                        cplx t = cmul(q12, q3);
                        s.re += t.re * c;
                    }
                }
            }
            nrm += s.re * s.re;
            T[(j*d2 + l)*d3 + m] = (float)s.re;
        }
    float invf = (float)(1.0 / sqrt(nrm));
    for (int e = 0; e < d1*d2*d3; ++e) T[e] *= invf;
}

// ---------------- pack q/k/v weights into WMMA B-operand f16 layout ----------------
// layout: wpk[(((tp*11+p)*8+kk)*32+lane)*16 + h], K32 = 16*(h>>3) + 8*(lane>>4) + (h&7), uv = kk*32+K32, n = lane&15
__global__ void pack_kernel(const float* __restrict__ wq, const float* __restrict__ wk,
                            const float* __restrict__ wv, _Float16* __restrict__ wpk){
    int gid = blockIdx.x * 256 + threadIdx.x;
    if (gid >= 3*NPATH*8*32*16) return;
    int h    = gid & 15;
    int lane = (gid >> 4) & 31;
    int kk   = (gid >> 9) & 7;
    int rest = gid >> 12;
    int pp   = rest % NPATH;
    int tp   = rest / NPATH;
    int l3   = cL3[pp];
    const float npaths[3] = {3.f, 4.f, 4.f};
    float scale = sqrtf((float)(2*l3+1) / (npaths[l3] * 256.f));
    int w   = lane & 15;
    int K32 = ((h >> 3) << 4) + ((lane >> 4) << 3) + (h & 7);     // K32 = 16*(h>=8) + 8*(lane>>4) + (h&7)
    int uv  = kk*32 + K32;
    int u = uv >> 4, v = uv & 15;
    const float* W = (tp == 0) ? wq : ((tp == 1) ? wk : wv);
    wpk[gid] = (_Float16)(scale * W[((pp*16 + u)*16 + v)*16 + w]);
}

// ---------------- main fused kernel: 16 nodes / block, 8 waves ----------------

__global__ void __launch_bounds__(256)
tfn_attn_kernel(const float* __restrict__ x, const float* __restrict__ wdot,
                const float* __restrict__ w3j, const _Float16* __restrict__ wpk,
                float* __restrict__ out, int N){
    __shared__ float    xs[TILE][FEAT];          // 9216 B
    __shared__ float    uabRaw[3*TILE*FEAT];     // 27648 B : a-buffer / out-buffer union
    __shared__ _Float16 bA[5*8*32*16];           // 40960 B : WMMA A-operand (f16)

    float (*aBuf)[5][16][5] = reinterpret_cast<float(*)[5][16][5]>(uabRaw);
    float* outb  = uabRaw;                        // [3][16][144], valid after path loop
    float* dpart = reinterpret_cast<float*>(bA);  // reuse bA after WMMA phase
    float* attnS = dpart + 256;

    const int tid  = threadIdx.x;
    const int lane = tid & 31;
    const int wave = tid >> 5;
    const int base0 = blockIdx.x * TILE;

    // load x tile into LDS
#ifdef USE_ASYNC_LDS
    for (int e = tid; e < (TILE*FEAT)/4; e += 256) {
        int fi = e * 4;
        int node = fi / FEAT, c = fi % FEAT;
        int g = base0 + node;
        if (g >= N) g = N - 1;                    // clamp; padded rows never written out
        __builtin_amdgcn_global_load_async_to_lds_b128(
            (gptr4)(x + (size_t)g*FEAT + c),
            (lptr4)&xs[node][c], 0, 0);
    }
    __builtin_amdgcn_s_wait_asynccnt(0);
#else
    for (int e = tid; e < TILE*FEAT; e += 256) {
        int node = e / FEAT, c = e % FEAT;
        int g = base0 + node;
        xs[node][c] = (g < N) ? x[g*FEAT + c] : 0.f;
    }
#endif
    if (tid == 0 && base0 + TILE < N)
        __builtin_prefetch(&x[(base0 + TILE)*FEAT], 0, 1);   // global_prefetch_b8
    __syncthreads();

    v8f acc[4];
    #pragma unroll
    for (int i = 0; i < 4; ++i) acc[i] = (v8f)0.f;

    const int nodeT = tid >> 4;
    const int laneU = tid & 15;

    for (int p = 0; p < NPATH; ++p) {
        const int l1 = cL1[p], l2 = cL2[p], l3 = cL3[p];
        const int d1 = 2*l1+1, d2 = 2*l2+1, d3 = 2*l3+1;
        const int o1 = cOFF[l1], o2 = cOFF[l2];
        const float* C = w3j + p * 128;

        // a[node][i][v][k] = sum_j x2[node,v,j] * C[i,j,k]   (thread = (node, v))
        for (int i = 0; i < d1; ++i)
            for (int k = 0; k < d3; ++k) {
                float s = 0.f;
                for (int j = 0; j < d2; ++j)
                    s += xs[nodeT][o2 + laneU*d2 + j] * C[(i*d2 + j)*d3 + k];
                aBuf[nodeT][i][laneU][k] = s;
            }
        __syncthreads();

        // b[node][u][v][k] = sum_i x1[node,u,i] * a[node][i][v][k], pack to WMMA A layout
        {
            float x1v[5];
            for (int i = 0; i < d1; ++i) x1v[i] = xs[nodeT][o1 + laneU*d1 + i];
            for (int v = 0; v < 16; v += 2) {
                const int uv  = laneU*16 + v;        // even
                const int kk  = uv >> 5;
                const int K32 = uv & 31;
                // 16-bit A-matrix 16x32 ISA layout: K32 = 16*(h>=8) + 8*(lane>>4) + (h&7)
                const int laneT = nodeT + (((K32 >> 3) & 1) << 4);
                const int h     = (((K32 >> 4) & 1) << 3) + (K32 & 7);
                for (int k = 0; k < d3; ++k) {
                    float b0 = 0.f, b1 = 0.f;
                    for (int i = 0; i < d1; ++i) {
                        b0 += x1v[i] * aBuf[nodeT][i][v  ][k];
                        b1 += x1v[i] * aBuf[nodeT][i][v+1][k];
                    }
                    union { _Float16 hh[2]; unsigned int u32; } pk2;
                    pk2.hh[0] = (_Float16)b0; pk2.hh[1] = (_Float16)b1;
                    *reinterpret_cast<unsigned int*>(&bA[(((k*8 + kk)*32 + laneT) << 4) + h]) = pk2.u32;
                }
            }
        }
        __syncthreads();

        // WMMA phase: wave owns output tiles t with t % 8 == wave; t = tp*9 + slice
        #pragma unroll
        for (int ord = 0; ord < 4; ++ord) {
            int t = wave + 8*ord;
            if (t >= 27) break;
            int tp = t / 9, slice = t % 9;
            int l3s = (slice == 0) ? 0 : ((slice < 4) ? 1 : 2);
            if (l3s != l3) continue;
            int kc = slice - ((l3s == 0) ? 0 : ((l3s == 1) ? 1 : 4));
            const _Float16* Bp = wpk + (size_t)((tp*NPATH + p) * 8) * 512;
            #pragma unroll
            for (int kk = 0; kk < 8; ++kk) {
                v16h Av = *reinterpret_cast<const v16h*>(&bA[(((kc*8 + kk)*32 + lane) << 4)]);
                v16h Bv = *reinterpret_cast<const v16h*>(&Bp[((kk*32 + lane) << 4)]);
                acc[ord] = __builtin_amdgcn_wmma_f32_16x16x32_f16(
                    false, Av, false, Bv, (short)0, acc[ord], false, false);
            }
        }
        __syncthreads();
    }

    // write accumulators -> outb[tp][node][144]
    #pragma unroll
    for (int ord = 0; ord < 4; ++ord) {
        int t = wave + 8*ord;
        if (t >= 27) break;
        int tp = t / 9, slice = t % 9;
        int l3s = (slice == 0) ? 0 : ((slice < 4) ? 1 : 2);
        int kc = slice - ((l3s == 0) ? 0 : ((l3s == 1) ? 1 : 4));
        int base = (l3s == 0) ? 0 : ((l3s == 1) ? 16 : 64);
        int dd = 2*l3s + 1;
        int wcol = lane & 15;
        int nb = (lane >> 4) << 3;
        #pragma unroll
        for (int r = 0; r < 8; ++r)
            outb[(tp*16 + (r + nb))*FEAT + base + wcol*dd + kc] = acc[ord][r];
    }
    __syncthreads();

    // dot(q, k, wdot): thread (node, u)
    {
        float s = 0.f;
        for (int l = 0; l < 3; ++l) {
            int dl = 2*l + 1;
            int base = cOFF[l];
            const float* Cd = w3j + l * 128;   // paths 0,1,2 are (l,l,0); table [i][j][0]
            for (int i = 0; i < dl; ++i) {
                float qv = outb[(0*16 + nodeT)*FEAT + base + laneU*dl + i];
                for (int j = 0; j < dl; ++j) {
                    float c = Cd[i*dl + j];
                    float t2 = 0.f;
                    for (int v = 0; v < 16; ++v)
                        t2 += wdot[(l*16 + laneU)*16 + v] * outb[(1*16 + nodeT)*FEAT + base + v*dl + j];
                    s += c * qv * t2;
                }
            }
        }
        dpart[tid] = s;
    }
    __syncthreads();
    if (tid < TILE) {
        float a2 = 0.f;
        for (int u = 0; u < 16; ++u) a2 += dpart[tid*16 + u];
        float d = a2 * 0.036084391824351613f;        // sqrt(1/768)
        float e = __expf(d * (1.f/12.f));            // / sqrt(144)
        attnS[tid] = e / (e + 1e-10f);
    }
    __syncthreads();

    // out = elu(attn * v), streamed with non-temporal stores
    for (int e = tid; e < TILE*FEAT; e += 256) {
        int node = e / FEAT, idx = e % FEAT;
        int g = base0 + node;
        if (g < N) {
            float val = attnS[node] * outb[(2*16 + node)*FEAT + idx];
            val = (val > 0.f) ? val : (__expf(val) - 1.f);
            __builtin_nontemporal_store(val, &out[(size_t)g*FEAT + idx]);
        }
    }
}

// ---------------- launch ----------------

extern "C" void kernel_launch(void* const* d_in, const int* in_sizes, int n_in,
                              void* d_out, int out_size, void* d_ws, size_t ws_size,
                              hipStream_t stream){
    const float* x  = (const float*)d_in[0];
    const float* wq = (const float*)d_in[4];
    const float* wk = (const float*)d_in[5];
    const float* wv = (const float*)d_in[6];
    const float* wd = (const float*)d_in[7];
    int N = in_sizes[0] / FEAT;

    float*    w3j = (float*)d_ws;                                   // 11 * 128 floats
    _Float16* wpk = (_Float16*)((char*)d_ws + 8192);                // 3*11*8*32*16 halves

    w3j_kernel<<<1, 16, 0, stream>>>(w3j);
    pack_kernel<<<(3*NPATH*8*32*16 + 255) / 256, 256, 0, stream>>>(wq, wk, wv, wpk);

    int nblocks = (N + TILE - 1) / TILE;
    tfn_attn_kernel<<<nblocks, 256, 0, stream>>>(x, wd, w3j, wpk, (float*)d_out, N);
}
